// ObjectOrientedAttentionNetwork_59966333387001
// MI455X (gfx1250) — compile-verified, matrix operand order checked by reading, not run on hardware
//
#include <hip/hip_runtime.h>

// ---------------- types ----------------
typedef __bf16 v16bf __attribute__((ext_vector_type(16)));
typedef __bf16 v8bf  __attribute__((ext_vector_type(8)));
typedef float  v8f   __attribute__((ext_vector_type(8)));

union Frag16 {
    v16bf v;
    v8bf  h[2];
    __bf16 e[16];
};

#define WMMA_BF16(a, b, c) \
    __builtin_amdgcn_wmma_f32_16x16x32_bf16(false, (a), false, (b), (short)0, (c), false, false)

#if defined(__has_builtin)
#if __has_builtin(__builtin_amdgcn_sched_barrier)
#define SCHED_FENCE() __builtin_amdgcn_sched_barrier(0)
#endif
#endif
#ifndef SCHED_FENCE
#define SCHED_FENCE() ((void)0)
#endif

// gfx1250 async global->LDS copy, 16B per lane, tracked by ASYNCcnt.
// Low 32 bits of a generic pointer into LDS are the LDS byte offset
__device__ __forceinline__ void async_copy_b128(const void* gaddr, void* lds_ptr) {
    unsigned lds_off = (unsigned)(unsigned long long)lds_ptr;
    asm volatile("global_load_async_to_lds_b128 %0, %1, off"
                 :: "v"(lds_off), "v"(gaddr)
                 : "memory");
}
__device__ __forceinline__ void wait_async0() {
    asm volatile("s_wait_asynccnt 0x0" ::: "memory");
}

// ---------------- problem constants ----------------
#define BSZ   128
#define LLEN  80
#define NVIS  36
#define DVIS  12544
#define DTXT  300
#define DHID  512
#define MTXT  (BSZ * LLEN)   // 10240
#define MIMG  (BSZ * NVIS)   // 4608
#define KPAD_TXT 1504
#define KPAD_CV  320
#define NPAD_CV  304

// ============================================================
// repack: fp32 -> bf16, with row/col zero padding and optional
// strided gather (for conv weights stored (O, I, k)).
// ============================================================
__global__ void repack_bf16(const float* __restrict__ src, __bf16* __restrict__ dst,
                            int rows_src, int rows_pad, int K, int Kpad,
                            int stride, int offset) {
    long i = (long)blockIdx.x * blockDim.x + threadIdx.x;
    long total = (long)rows_pad * Kpad;
    if (i >= total) return;
    int r = (int)(i / Kpad);
    int k = (int)(i % Kpad);
    float v = 0.0f;
    if (r < rows_src && k < K) v = src[((long)r * K + k) * stride + offset];
    dst[i] = (__bf16)v;
}

// zero the 4 pad columns (1500..1503) of xcat
__global__ void zero_xcat_pad(__bf16* __restrict__ xcat) {
    int i = blockIdx.x * blockDim.x + threadIdx.x;
    if (i < MTXT * 4) {
        int r = i >> 2, c = 1500 + (i & 3);
        xcat[(long)r * KPAD_TXT + c] = (__bf16)0.0f;
    }
}

// ============================================================
// Generic bf16 WMMA GEMM:  C(M,N) = act( A(M,K) * B^T + bias )
//   A: fp32 (VALU convert while staging) or bf16 (async copy to LDS)
//   B: bf16 "B^T" layout (row n holds K contiguous), async copy to LDS
//   block tile 128M x 128N, 8 waves, each wave 16M x 128N
//   double-buffered LDS: async stage k+32 overlapped with WMMA on k
//   sched_barrier pins all fragment ds_loads before the WMMA burst
//   requires M%128==0, N%128==0, K%32==0
// ============================================================
__global__ __launch_bounds__(256)
void gemm_bf16_wmma(const void* __restrict__ Av, int a_is_f32, int lda,
                    const __bf16* __restrict__ Bw, int ldb,
                    const float* __restrict__ bias, int act_tanh,
                    float* __restrict__ outf, __bf16* __restrict__ outb,
                    int ldc, int K) {
    __shared__ __bf16 smA[2][128 * 40];
    __shared__ __bf16 smB[2][128 * 40];

    const int tid  = threadIdx.x;
    const int lane = tid & 31;
    const int wid  = tid >> 5;
    const int r    = lane & 15;
    const int hs   = lane >> 4;      // half-select
    const int kb   = hs * 8;         // K base within fragment per ISA layout

    const long tileM = (long)blockIdx.x * 128;
    const long tileN = (long)blockIdx.y * 128;

    v8f acc[8];
#pragma unroll
    for (int i = 0; i < 8; ++i) acc[i] = (v8f){0.f,0.f,0.f,0.f,0.f,0.f,0.f,0.f};

    const int srow = tid >> 1;          // staging row 0..127
    const int skk  = (tid & 1) * 16;    // staging k sub-chunk (0 or 16)

    auto stageA = [&](int buf, long k0) {
        if (a_is_f32) {
            const float* ap = (const float*)Av + (tileM + srow) * lda + k0 + skk;
            v8bf h0, h1;
#pragma unroll
            for (int i = 0; i < 8; ++i) h0[i] = (__bf16)ap[i];
#pragma unroll
            for (int i = 0; i < 8; ++i) h1[i] = (__bf16)ap[8 + i];
            *(v8bf*)&smA[buf][srow * 40 + skk]     = h0;
            *(v8bf*)&smA[buf][srow * 40 + skk + 8] = h1;
            if (k0 + 32 < K) __builtin_prefetch(ap + 32, 0, 1);
        } else {
            const __bf16* ap = (const __bf16*)Av + (tileM + srow) * lda + k0 + skk;
            async_copy_b128(ap,     &smA[buf][srow * 40 + skk]);
            async_copy_b128(ap + 8, &smA[buf][srow * 40 + skk + 8]);
        }
    };
    auto stageB = [&](int buf, long k0) {
        const __bf16* bp = Bw + (tileN + srow) * (long)ldb + k0 + skk;
        async_copy_b128(bp,     &smB[buf][srow * 40 + skk]);
        async_copy_b128(bp + 8, &smB[buf][srow * 40 + skk + 8]);
    };

    // prologue: stage first tile
    stageA(0, 0);
    stageB(0, 0);
    wait_async0();
    __syncthreads();

    int cur = 0;
    for (long k0 = 0; k0 < K; k0 += 32) {
        const long kn = k0 + 32;
        // stage next tile into the other buffer (overlaps with WMMA below)
        if (kn < K) {
            stageA(cur ^ 1, kn);
            stageB(cur ^ 1, kn);
        }

        // ---- load A fragment + all 8 B fragments ----
        Frag16 fa;
        const __bf16* pa = &smA[cur][(wid * 16 + r) * 40 + kb];
        fa.h[0] = *(const v8bf*)pa;          // K = kb .. kb+7
        fa.h[1] = *(const v8bf*)(pa + 16);   // K = kb+16 .. kb+23
        v16bf fb[8];
#pragma unroll
        for (int nt = 0; nt < 8; ++nt) {
            Frag16 t;
            const __bf16* pb = &smB[cur][(nt * 16 + r) * 40 + kb];
            t.h[0] = *(const v8bf*)pb;
            t.h[1] = *(const v8bf*)(pb + 16);
            fb[nt] = t.v;
        }
        // keep all ds_loads above, all WMMAs below: one dscnt wait, then
        // an uninterrupted back-to-back WMMA burst (no inter-WMMA hazards:
        // accumulators are disjoint)
        SCHED_FENCE();
#pragma unroll
        for (int nt = 0; nt < 8; ++nt)
            acc[nt] = WMMA_BF16(fa.v, fb[nt], acc[nt]);

        if (kn < K) {
            wait_async0();       // our async stage of buffer cur^1 done
            __syncthreads();     // everyone's stage done / reads of cur done
            cur ^= 1;
        }
    }

    // ---- epilogue: bias + activation + stores ----
#pragma unroll
    for (int nt = 0; nt < 8; ++nt) {
        long col = tileN + nt * 16 + r;
        float bv = bias ? bias[col] : 0.0f;
#pragma unroll
        for (int rr = 0; rr < 8; ++rr) {
            long row = tileM + wid * 16 + (hs ? rr + 8 : rr);
            float val = acc[nt][rr] + bv;
            if (act_tanh) val = tanhf(val);
            if (outf) outf[row * ldc + col] = val;
            if (outb) outb[row * ldc + col] = (__bf16)val;
        }
    }
}

// ============================================================
// Conv (k=1,2,3,7, replication padding) as per-tap WMMA GEMMs.
// ============================================================
__global__ __launch_bounds__(256)
void conv_wmma(const float* __restrict__ txts, const __bf16* __restrict__ convW,
               const float* __restrict__ b1, const float* __restrict__ b2,
               const float* __restrict__ b3, const float* __restrict__ b7,
               __bf16* __restrict__ xcat) {
    const int ks[4] = {1, 2, 3, 7};
    const int pl[4] = {0, 0, 1, 3};
    const int tb[4] = {0, 1, 3, 6};      // tap block base index
    const int cb[4] = {0, 300, 600, 1200};

    const int conv = blockIdx.z;
    const int KS = ks[conv], P = pl[conv];
    const float* bias = (conv == 0) ? b1 : (conv == 1) ? b2 : (conv == 2) ? b3 : b7;

    const int lane = threadIdx.x & 31;
    const int wid  = threadIdx.x >> 5;
    const int r    = lane & 15;
    const int hs   = lane >> 4;
    const int kb   = hs * 8;

    const int m  = blockIdx.x * 128 + wid * 16 + r;   // A-row for this lane
    const int bb = m / LLEN;
    const int pos = m % LLEN;
    const int n  = blockIdx.y * 16 + r;               // output channel

    v8f acc = (v8f){0.f,0.f,0.f,0.f,0.f,0.f,0.f,0.f};

    for (int d = 0; d < KS; ++d) {
        int sp = pos + d - P;
        sp = sp < 0 ? 0 : (sp > LLEN - 1 ? LLEN - 1 : sp);
        const float* arow = txts + ((long)bb * LLEN + sp) * DTXT;
        const __bf16* brow = convW + (long)(tb[conv] + d) * NPAD_CV * KPAD_CV
                                   + (long)n * KPAD_CV;
        for (int k0 = 0; k0 < KPAD_CV; k0 += 32) {
            Frag16 fa;
#pragma unroll
            for (int i = 0; i < 8; ++i) {
                int k = k0 + kb + i;
                fa.e[i] = (k < DTXT) ? (__bf16)arow[k] : (__bf16)0.0f;
            }
#pragma unroll
            for (int i = 0; i < 8; ++i) {
                int k = k0 + kb + 16 + i;
                fa.e[8 + i] = (k < DTXT) ? (__bf16)arow[k] : (__bf16)0.0f;
            }
            Frag16 fb;
            fb.h[0] = *(const v8bf*)(brow + k0 + kb);
            fb.h[1] = *(const v8bf*)(brow + k0 + kb + 16);
            SCHED_FENCE();
            acc = WMMA_BF16(fa.v, fb.v, acc);
        }
    }

    if (n < DTXT) {
        float bv = bias[n];
#pragma unroll
        for (int rr = 0; rr < 8; ++rr) {
            long row = (long)blockIdx.x * 128 + wid * 16 + (hs ? rr + 8 : rr);
            float val = tanhf(acc[rr] + bv);
            __bf16 h = (__bf16)val;
            xcat[row * KPAD_TXT + cb[conv] + n] = h;
            if (conv == 2) xcat[row * KPAD_TXT + 900 + n] = h;  // x5 = x3
        }
    }
}

// ============================================================
// small fp32 helper kernels
// ============================================================
__global__ void row_norm_k(const float* __restrict__ x, float* __restrict__ out, int rows) {
    int wid = threadIdx.x >> 5, lane = threadIdx.x & 31;
    int row = blockIdx.x * 8 + wid;
    if (row >= rows) return;
    const float* p = x + (long)row * DHID;
    float s = 0.f;
    for (int i = lane; i < DHID; i += 32) { float v = p[i]; s += v * v; }
    for (int off = 16; off; off >>= 1) s += __shfl_xor(s, off, 32);
    if (lane == 0) out[row] = sqrtf(s);
}

__global__ void col_mean_k(const float* __restrict__ x, float* __restrict__ out, int R) {
    int b = blockIdx.x;
    for (int d = threadIdx.x; d < DHID; d += blockDim.x) {
        float s = 0.f;
        for (int r = 0; r < R; ++r) s += x[((long)b * R + r) * DHID + d];
        out[(long)b * DHID + d] = s / (float)R;
    }
}

__global__ void sim_k(const float* __restrict__ t, const float* __restrict__ v,
                      const float* __restrict__ tn, const float* __restrict__ vn,
                      float* __restrict__ sim) {
    int b = blockIdx.x;
    int p = blockIdx.y * blockDim.x + threadIdx.x;
    if (p >= LLEN * NVIS) return;
    int l = p / NVIS, vv = p % NVIS;
    const float4* tp = (const float4*)(t + ((long)b * LLEN + l) * DHID);
    const float4* vp = (const float4*)(v + ((long)b * NVIS + vv) * DHID);
    float s = 0.f;
    for (int i = 0; i < DHID / 4; ++i) {
        float4 a = tp[i], c = vp[i];
        s += a.x * c.x + a.y * c.y + a.z * c.z + a.w * c.w;
    }
    float den = fmaxf(tn[b * LLEN + l] * vn[b * NVIS + vv], 1e-8f);
    sim[(long)b * (LLEN * NVIS) + p] = s / den;
}

// cross over V axis (row length 36): relu -> L1 norm -> softmax(x*9)
__global__ void cross_v2t_k(const float* __restrict__ sim, float* __restrict__ w) {
    int wid = threadIdx.x >> 5, lane = threadIdx.x & 31;
    long row = (long)blockIdx.x * 8 + wid;
    if (row >= MTXT) return;
    const float* s = sim + row * NVIS;
    float a0 = fmaxf(s[lane], 0.f);
    float a1 = (lane < 4) ? fmaxf(s[32 + lane], 0.f) : 0.f;
    float sm = a0 + a1;
    for (int off = 16; off; off >>= 1) sm += __shfl_xor(sm, off, 32);
    float sc = 9.0f / fmaxf(sm, 1e-10f);
    float x0 = a0 * sc;
    float x1 = a1 * sc;
    float mx = fmaxf(x0, (lane < 4) ? x1 : -1e30f);
    for (int off = 16; off; off >>= 1) mx = fmaxf(mx, __shfl_xor(mx, off, 32));
    float e0 = __expf(x0 - mx);
    float e1 = (lane < 4) ? __expf(x1 - mx) : 0.f;
    float es = e0 + e1;
    for (int off = 16; off; off >>= 1) es += __shfl_xor(es, off, 32);
    float inv = 1.0f / es;
    w[row * NVIS + lane] = e0 * inv;
    if (lane < 4) w[row * NVIS + 32 + lane] = e1 * inv;
}

// cross over L axis (row length 80, strided reads of sim^T)
__global__ void cross_t2v_k(const float* __restrict__ sim, float* __restrict__ w) {
    int wid = threadIdx.x >> 5, lane = threadIdx.x & 31;
    long row = (long)blockIdx.x * 8 + wid;
    if (row >= MIMG) return;
    int b = (int)(row / NVIS), vv = (int)(row % NVIS);
    const float* s = sim + (long)b * (LLEN * NVIS) + vv;   // stride NVIS over l
    float a[3];
#pragma unroll
    for (int j = 0; j < 3; ++j) {
        int l = lane + j * 32;
        a[j] = (l < LLEN) ? fmaxf(s[(long)l * NVIS], 0.f) : 0.f;
    }
    float sm = a[0] + a[1] + a[2];
    for (int off = 16; off; off >>= 1) sm += __shfl_xor(sm, off, 32);
    float sc = 9.0f / fmaxf(sm, 1e-10f);
    float x[3], mx = -1e30f;
#pragma unroll
    for (int j = 0; j < 3; ++j) {
        int l = lane + j * 32;
        x[j] = a[j] * sc;
        if (l < LLEN) mx = fmaxf(mx, x[j]);
    }
    for (int off = 16; off; off >>= 1) mx = fmaxf(mx, __shfl_xor(mx, off, 32));
    float e[3], es = 0.f;
#pragma unroll
    for (int j = 0; j < 3; ++j) {
        int l = lane + j * 32;
        e[j] = (l < LLEN) ? __expf(x[j] - mx) : 0.f;
        es += e[j];
    }
    for (int off = 16; off; off >>= 1) es += __shfl_xor(es, off, 32);
    float inv = 1.0f / es;
#pragma unroll
    for (int j = 0; j < 3; ++j) {
        int l = lane + j * 32;
        if (l < LLEN) w[row * LLEN + l] = e[j] * inv;
    }
}

// visual_attended_text[b,v,d] = sum_l w_v2t[b,l,v] * t[b,l,d]
__global__ void vat_k(const float* __restrict__ w, const float* __restrict__ t,
                      float* __restrict__ out) {
    __shared__ float sw[LLEN];
    int b = blockIdx.x, vv = blockIdx.y;
    if (threadIdx.x < LLEN) sw[threadIdx.x] = w[((long)b * LLEN + threadIdx.x) * NVIS + vv];
    __syncthreads();
    for (int d = threadIdx.x; d < DHID; d += blockDim.x) {
        float s = 0.f;
        for (int l = 0; l < LLEN; ++l) s += sw[l] * t[((long)b * LLEN + l) * DHID + d];
        out[((long)b * NVIS + vv) * DHID + d] = s;
    }
}

// textual_attended_vision[b,l,d] = sum_v w_t2v[b,v,l] * v[b,v,d]
__global__ void tav_k(const float* __restrict__ w, const float* __restrict__ v,
                      float* __restrict__ out) {
    __shared__ float sw[NVIS];
    int b = blockIdx.x, l = blockIdx.y;
    if (threadIdx.x < NVIS) sw[threadIdx.x] = w[((long)b * NVIS + threadIdx.x) * LLEN + l];
    __syncthreads();
    for (int d = threadIdx.x; d < DHID; d += blockDim.x) {
        float s = 0.f;
        for (int vv = 0; vv < NVIS; ++vv) s += sw[vv] * v[((long)b * NVIS + vv) * DHID + d];
        out[((long)b * LLEN + l) * DHID + d] = s;
    }
}

// intra-attention (text R=80 / vision R=36); only q<6 needed.
__global__ void intra_att_k(const float* __restrict__ src, int R,
                            const float* __restrict__ mean,
                            const float* __restrict__ cW, const float* __restrict__ cB,
                            const __bf16* __restrict__ c2, float* __restrict__ out) {
    __shared__ float c1[DHID];
    __shared__ float lg[LLEN];
    __shared__ float sinv;
    int b = blockIdx.x, q = blockIdx.y;
    float cw = cW[q], cbv = cB[q];
    for (int d = threadIdx.x; d < DHID; d += blockDim.x)
        c1[d] = tanhf(mean[(long)b * DHID + d] * cw + cbv);
    __syncthreads();
    int wid = threadIdx.x >> 5, lane = threadIdx.x & 31;
    for (int r = wid; r < R; r += 8) {
        const __bf16* p = c2 + ((long)b * R + r) * DHID;
        float s = 0.f;
        for (int d = lane; d < DHID; d += 32) s += c1[d] * (float)p[d];
        for (int off = 16; off; off >>= 1) s += __shfl_xor(s, off, 32);
        if (lane == 0) lg[r] = s * 9.0f;
    }
    __syncthreads();
    if (threadIdx.x == 0) {
        float m = -1e30f;
        for (int r = 0; r < R; ++r) m = fmaxf(m, lg[r]);
        float ss = 0.f;
        for (int r = 0; r < R; ++r) { float e = __expf(lg[r] - m); lg[r] = e; ss += e; }
        sinv = 1.0f / ss;
    }
    __syncthreads();
    float inv = sinv;
    for (int d = threadIdx.x; d < DHID; d += blockDim.x) {
        float s = 0.f;
        for (int r = 0; r < R; ++r) s += lg[r] * src[((long)b * R + r) * DHID + d];
        out[((long)b * 6 + q) * DHID + d] = s * inv;
    }
}

// ============================================================
// launch
// ============================================================
extern "C" void kernel_launch(void* const* d_in, const int* in_sizes, int n_in,
                              void* d_out, int out_size, void* d_ws, size_t ws_size,
                              hipStream_t stream) {
    const float* txts    = (const float*)d_in[0];
    const float* detect  = (const float*)d_in[1];
    const float* img_W   = (const float*)d_in[2];
    const float* img_b   = (const float*)d_in[3];
    const float* conv1_W = (const float*)d_in[4];
    const float* conv1_b = (const float*)d_in[5];
    const float* conv2_W = (const float*)d_in[6];
    const float* conv2_b = (const float*)d_in[7];
    const float* conv3_W = (const float*)d_in[8];
    const float* conv3_b = (const float*)d_in[9];
    const float* conv7_W = (const float*)d_in[10];
    const float* conv7_b = (const float*)d_in[11];
    const float* txt_W   = (const float*)d_in[12];
    const float* txt_b   = (const float*)d_in[13];
    const float* ct_W    = (const float*)d_in[14];
    const float* ct_b    = (const float*)d_in[15];
    const float* t_W     = (const float*)d_in[16];
    const float* t_b     = (const float*)d_in[17];
    const float* cv_W    = (const float*)d_in[18];
    const float* cv_b    = (const float*)d_in[19];
    const float* v_W     = (const float*)d_in[20];
    const float* v_b     = (const float*)d_in[21];

    float* out = (float*)d_out;
    const long O_V     = 0;
    const long O_T     = O_V + (long)MIMG * DHID;      //  2359296
    const long O_VAT   = O_T + (long)MTXT * DHID;      //  7602176
    const long O_TAV   = O_VAT + (long)MIMG * DHID;    //  9961472
    const long O_ATT_T = O_TAV + (long)MTXT * DHID;    // 15204352
    const long O_ATT_V = O_ATT_T + (long)BSZ * 6 * DHID;

    // ---- workspace carve-up ----
    char* ws = (char*)d_ws;
    size_t off = 0;
    auto take = [&](size_t bytes) -> void* {
        void* p = ws + off;
        off = (off + bytes + 255) & ~(size_t)255;
        return p;
    };
    __bf16* imgWb  = (__bf16*)take((size_t)DHID * DVIS * 2);
    __bf16* txtWb  = (__bf16*)take((size_t)DHID * KPAD_TXT * 2);
    __bf16* tWb    = (__bf16*)take((size_t)DHID * DHID * 2);
    __bf16* vWb    = (__bf16*)take((size_t)DHID * DHID * 2);
    __bf16* convWb = (__bf16*)take((size_t)13 * NPAD_CV * KPAD_CV * 2);
    __bf16* xcat   = (__bf16*)take((size_t)MTXT * KPAD_TXT * 2);
    __bf16* tbf    = (__bf16*)take((size_t)MTXT * DHID * 2);
    __bf16* vbf    = (__bf16*)take((size_t)MIMG * DHID * 2);
    __bf16* cT2    = (__bf16*)take((size_t)MTXT * DHID * 2);
    __bf16* cV2    = (__bf16*)take((size_t)MIMG * DHID * 2);
    float*  tmean  = (float*)take((size_t)BSZ * DHID * 4);
    float*  vmean  = (float*)take((size_t)BSZ * DHID * 4);
    float*  tn     = (float*)take((size_t)MTXT * 4);
    float*  vn     = (float*)take((size_t)MIMG * 4);
    float*  simb   = (float*)take((size_t)BSZ * LLEN * NVIS * 4);
    float*  wv2t   = (float*)take((size_t)BSZ * LLEN * NVIS * 4);
    float*  wt2v   = (float*)take((size_t)BSZ * NVIS * LLEN * 4);

    auto rblocks = [](long total) { return (unsigned)((total + 255) / 256); };

    // ---- weight repacks (fp32 -> bf16, padded) ----
    repack_bf16<<<rblocks((long)DHID * DVIS), 256, 0, stream>>>(img_W, imgWb, DHID, DHID, DVIS, DVIS, 1, 0);
    repack_bf16<<<rblocks((long)DHID * KPAD_TXT), 256, 0, stream>>>(txt_W, txtWb, DHID, DHID, 5 * DTXT, KPAD_TXT, 1, 0);
    repack_bf16<<<rblocks((long)DHID * DHID), 256, 0, stream>>>(t_W, tWb, DHID, DHID, DHID, DHID, 1, 0);
    repack_bf16<<<rblocks((long)DHID * DHID), 256, 0, stream>>>(v_W, vWb, DHID, DHID, DHID, DHID, 1, 0);
    {
        const float* cw[4] = {conv1_W, conv2_W, conv3_W, conv7_W};
        const int ks[4] = {1, 2, 3, 7};
        int tap = 0;
        for (int c = 0; c < 4; ++c)
            for (int d = 0; d < ks[c]; ++d, ++tap)
                repack_bf16<<<rblocks((long)NPAD_CV * KPAD_CV), 256, 0, stream>>>(
                    cw[c], convWb + (long)tap * NPAD_CV * KPAD_CV,
                    DTXT, NPAD_CV, DTXT, KPAD_CV, ks[c], d);
    }

    // ---- ImgNet: v = detect @ img_W^T + b  (fp32 A converted on the fly) ----
    gemm_bf16_wmma<<<dim3(MIMG / 128, DHID / 128), 256, 0, stream>>>(
        detect, 1, DVIS, imgWb, DVIS, img_b, 0, out + O_V, vbf, DHID, DVIS);

    // ---- TxtNet convs -> xcat (bf16, 1504-wide) ----
    conv_wmma<<<dim3(MTXT / 128, (NPAD_CV / 16), 4), 256, 0, stream>>>(
        txts, convWb, conv1_b, conv2_b, conv3_b, conv7_b, xcat);
    zero_xcat_pad<<<rblocks((long)MTXT * 4), 256, 0, stream>>>(xcat);

    // ---- t = tanh(xcat @ txt_W^T + b) ----
    gemm_bf16_wmma<<<dim3(MTXT / 128, DHID / 128), 256, 0, stream>>>(
        xcat, 0, KPAD_TXT, txtWb, KPAD_TXT, txt_b, 1, out + O_T, tbf, DHID, KPAD_TXT);

    // ---- c_T2 = tanh(t @ t_W^T + b), c_V2 = tanh(v @ v_W^T + b) ----
    gemm_bf16_wmma<<<dim3(MTXT / 128, DHID / 128), 256, 0, stream>>>(
        tbf, 0, DHID, tWb, DHID, t_b, 1, (float*)nullptr, cT2, DHID, DHID);
    gemm_bf16_wmma<<<dim3(MIMG / 128, DHID / 128), 256, 0, stream>>>(
        vbf, 0, DHID, vWb, DHID, v_b, 1, (float*)nullptr, cV2, DHID, DHID);

    // ---- norms, means ----
    row_norm_k<<<(MTXT + 7) / 8, 256, 0, stream>>>(out + O_T, tn, MTXT);
    row_norm_k<<<(MIMG + 7) / 8, 256, 0, stream>>>(out + O_V, vn, MIMG);
    col_mean_k<<<BSZ, 256, 0, stream>>>(out + O_T, tmean, LLEN);
    col_mean_k<<<BSZ, 256, 0, stream>>>(out + O_V, vmean, NVIS);

    // ---- cosine sim + cross attention ----
    sim_k<<<dim3(BSZ, (LLEN * NVIS + 255) / 256), 256, 0, stream>>>(
        out + O_T, out + O_V, tn, vn, simb);
    cross_v2t_k<<<(MTXT + 7) / 8, 256, 0, stream>>>(simb, wv2t);
    cross_t2v_k<<<(MIMG + 7) / 8, 256, 0, stream>>>(simb, wt2v);
    vat_k<<<dim3(BSZ, NVIS), 256, 0, stream>>>(wv2t, out + O_T, out + O_VAT);
    tav_k<<<dim3(BSZ, LLEN), 256, 0, stream>>>(wt2v, out + O_V, out + O_TAV);

    // ---- intra attention (only q<6 contributes to outputs) ----
    intra_att_k<<<dim3(BSZ, 6), 256, 0, stream>>>(
        out + O_T, LLEN, tmean, ct_W, ct_b, cT2, out + O_ATT_T);
    intra_att_k<<<dim3(BSZ, 6), 256, 0, stream>>>(
        out + O_V, NVIS, vmean, cv_W, cv_b, cV2, out + O_ATT_V);

    (void)in_sizes; (void)n_in; (void)out_size; (void)ws_size;
}
